// DCCLoss_7284264534291
// MI455X (gfx1250) — matrix operand order repaired and checked by reading, not compiled, not run on hardware
//
#include <hip/hip_runtime.h>
#include <hip/hip_bf16.h>
#include <cfloat>

typedef __attribute__((ext_vector_type(16))) __bf16 v16bf;
typedef __attribute__((ext_vector_type(8)))  __bf16 v8bf;
typedef __attribute__((ext_vector_type(4)))  __bf16 v4bf;
typedef __attribute__((ext_vector_type(8)))  float  v8f;

#define SCALARV 20.0f
#define LS_EPS  0.1f
#define WEIGHTV 0.25f
#define BATCH   128
#define FDIM    256
#define KCHUNKS (FDIM / 32)   // 8
#define MTILES  (BATCH / 16)  // 8
#define NEG_BIG (-3.0e38f)
#define AELEMS  (BATCH * FDIM)

// split fp32 -> bf16 hi + bf16 lo correction (a ~= hi + lo to ~2^-16 rel)
__device__ inline void cvt_split16(const float* __restrict__ t, v16bf& hi, v16bf& lo) {
#pragma unroll
  for (int i = 0; i < 16; ++i) {
    float a = t[i];
    __bf16 h = (__bf16)a;
    hi[i] = h;
    lo[i] = (__bf16)(a - (float)h);
  }
}

// B fragment gather: 16 contiguous floats (64B aligned)
__device__ inline void loadB16(const float* __restrict__ base, float* __restrict__ t) {
  const float4* q = (const float4*)base;
  float4 a = q[0], b = q[1], c = q[2], d = q[3];
  t[0]=a.x;  t[1]=a.y;  t[2]=a.z;  t[3]=a.w;
  t[4]=b.x;  t[5]=b.y;  t[6]=b.z;  t[7]=b.w;
  t[8]=c.x;  t[9]=c.y;  t[10]=c.z; t[11]=c.w;
  t[12]=d.x; t[13]=d.y; t[14]=d.z; t[15]=d.w;
}

// A fragment from LDS: two 16B chunks (8 bf16 each) -> v16bf
__device__ inline v16bf loadA_lds(const __bf16* __restrict__ plane, int row, int f) {
  const v8bf* p0 = (const v8bf*)(plane + row * FDIM + f);
  const v8bf* p1 = (const v8bf*)(plane + row * FDIM + f + 8);
  v8bf h0 = *p0, h1 = *p1;
  return __builtin_shufflevector(h0, h1, 0,1,2,3,4,5,6,7,8,9,10,11,12,13,14,15);
}

__global__ __launch_bounds__(256)
void dcc_gemm_partial(const float* __restrict__ inputs, const int* __restrict__ targets,
                      const float* __restrict__ lut_ccc, const float* __restrict__ lut_icc,
                      int C, int NG,
                      float* __restrict__ part_m_c, float* __restrict__ part_s_c, float* __restrict__ part_z_c,
                      float* __restrict__ part_m_i, float* __restrict__ part_s_i, float* __restrict__ part_z_i,
                      float* __restrict__ part_h,
                      float* __restrict__ zt_c, float* __restrict__ zt_i) {
  // dynamic LDS: A matrix as bf16 hi plane + lo plane (64KB + 64KB)
  extern __shared__ __bf16 smem_a[];
  __bf16* a_hi = smem_a;
  __bf16* a_lo = smem_a + AELEMS;

  __shared__ float lm_c[8][BATCH], ls_c[8][BATCH], lz_c[8][BATCH];
  __shared__ float lm_i[8][BATCH], ls_i[8][BATCH], lz_i[8][BATCH];
  __shared__ float lh[8];
  __shared__ int   lt[BATCH];

  const int tid  = threadIdx.x;
  const int lane = tid & 31;
  const int w    = tid >> 5;
  if (tid < BATCH) lt[tid] = targets[tid];

  // Cooperative one-time fp32 -> bf16(hi)/bf16(lo) conversion of A into LDS.
  for (int idx = tid * 4; idx < AELEMS; idx += 256 * 4) {
    float4 a = *(const float4*)(inputs + idx);
    v4bf h, l;
    h[0] = (__bf16)a.x; l[0] = (__bf16)(a.x - (float)h[0]);
    h[1] = (__bf16)a.y; l[1] = (__bf16)(a.y - (float)h[1]);
    h[2] = (__bf16)a.z; l[2] = (__bf16)(a.z - (float)h[2]);
    h[3] = (__bf16)a.w; l[3] = (__bf16)(a.w - (float)h[3]);
    *(v4bf*)(a_hi + idx) = h;
    *(v4bf*)(a_lo + idx) = l;
  }
  __syncthreads();

  const int g = blockIdx.x * 8 + w;           // 16-class group id
  if (g < NG) {
    const int  c0     = g * 16;
    const int  cls    = c0 + (lane & 15);     // B/C/D column for this lane
    const bool cvalid = (cls < C);
    const int  offB   = (lane & 16) ? 16 : 0; // B frag: K half per lane group
    const int  offA   = (lane & 16) ? 16 : 0; // A frag: K offset per lane group

    v8f acc_c[MTILES], acc_i[MTILES];
#pragma unroll
    for (int m = 0; m < MTILES; ++m) { v8f z = {}; acc_c[m] = z; acc_i[m] = z; }

    const float* bc_base = lut_ccc + (size_t)cls * FDIM + offB;
    const float* bi_base = lut_icc + (size_t)cls * FDIM + offB;

    for (int kc = 0; kc < KCHUNKS; ++kc) {
      const int f0 = kc * 32;
      float tb[16];
      v16bf bch, bcl, bih, bil;
      if (cvalid) {
        loadB16(bc_base + f0, tb);
      } else {
#pragma unroll
        for (int i = 0; i < 16; ++i) tb[i] = 0.f;
      }
      cvt_split16(tb, bch, bcl);
      if (cvalid) {
        loadB16(bi_base + f0, tb);
      } else {
#pragma unroll
        for (int i = 0; i < 16; ++i) tb[i] = 0.f;
      }
      cvt_split16(tb, bih, bil);

#pragma unroll
      for (int m = 0; m < MTILES; ++m) {
        const int row = m * 16 + (lane & 15);   // A row for this lane
        v16bf ah = loadA_lds(a_hi, row, f0 + offA);
        v16bf al = loadA_lds(a_lo, row, f0 + offA);
        // bf16x2 split product: hi*hi + hi*lo + lo*hi, fp32 accumulate
        acc_c[m] = __builtin_amdgcn_wmma_f32_16x16x32_bf16(false, ah, false, bch, (short)0, acc_c[m], false, false);
        acc_c[m] = __builtin_amdgcn_wmma_f32_16x16x32_bf16(false, ah, false, bcl, (short)0, acc_c[m], false, false);
        acc_c[m] = __builtin_amdgcn_wmma_f32_16x16x32_bf16(false, al, false, bch, (short)0, acc_c[m], false, false);
        acc_i[m] = __builtin_amdgcn_wmma_f32_16x16x32_bf16(false, ah, false, bih, (short)0, acc_i[m], false, false);
        acc_i[m] = __builtin_amdgcn_wmma_f32_16x16x32_bf16(false, ah, false, bil, (short)0, acc_i[m], false, false);
        acc_i[m] = __builtin_amdgcn_wmma_f32_16x16x32_bf16(false, al, false, bih, (short)0, acc_i[m], false, false);
      }
    }

    // Per-row partial stats over this wave's 16 classes.
    // C/D layout: row = 16*m + j + 8*(lane>=16), col = lane&15.
    float hsum = 0.f;
#pragma unroll
    for (int m = 0; m < MTILES; ++m) {
#pragma unroll
      for (int j = 0; j < 8; ++j) {
        const int row = m * 16 + j + ((lane & 16) ? 8 : 0);
        const float zc = acc_c[m][j] * SCALARV;
        const float zi = acc_i[m][j] * SCALARV;
        if (cvalid && cls == lt[row]) { zt_c[row] = zc; zt_i[row] = zi; }  // unique writer
        const float d  = fabsf(zc - zi);
        const float hu = (d < 1.f) ? (0.5f * d * d) : (d - 0.5f);
        hsum += cvalid ? hu : 0.f;

        float mc = cvalid ? zc : NEG_BIG;
        float mi = cvalid ? zi : NEG_BIG;
#pragma unroll
        for (int msk = 1; msk < 16; msk <<= 1) {   // reduce within 16-lane half
          mc = fmaxf(mc, __shfl_xor(mc, msk, 32));
          mi = fmaxf(mi, __shfl_xor(mi, msk, 32));
        }
        float ec = cvalid ? expf(zc - mc) : 0.f;
        float ei = cvalid ? expf(zi - mi) : 0.f;
        float sc = cvalid ? zc : 0.f;
        float si = cvalid ? zi : 0.f;
#pragma unroll
        for (int msk = 1; msk < 16; msk <<= 1) {
          ec += __shfl_xor(ec, msk, 32);
          ei += __shfl_xor(ei, msk, 32);
          sc += __shfl_xor(sc, msk, 32);
          si += __shfl_xor(si, msk, 32);
        }
        if ((lane & 15) == 0) {
          lm_c[w][row] = mc; ls_c[w][row] = ec; lz_c[w][row] = sc;
          lm_i[w][row] = mi; ls_i[w][row] = ei; lz_i[w][row] = si;
        }
      }
    }
#pragma unroll
    for (int msk = 1; msk < 32; msk <<= 1) hsum += __shfl_xor(hsum, msk, 32);
    if (lane == 0) lh[w] = hsum;
  } else {
    for (int r = lane; r < BATCH; r += 32) {
      lm_c[w][r] = NEG_BIG; ls_c[w][r] = 0.f; lz_c[w][r] = 0.f;
      lm_i[w][r] = NEG_BIG; ls_i[w][r] = 0.f; lz_i[w][r] = 0.f;
    }
    if (lane == 0) lh[w] = 0.f;
  }
  __syncthreads();

  // Deterministic in-block combine across the 8 waves, write per-block partials.
  const size_t bo = (size_t)blockIdx.x * BATCH;
  if (tid < BATCH) {
    const int r = tid;
    float M = NEG_BIG;
#pragma unroll
    for (int q = 0; q < 8; ++q) M = fmaxf(M, lm_c[q][r]);
    float S = 0.f, Z = 0.f;
#pragma unroll
    for (int q = 0; q < 8; ++q) { S += ls_c[q][r] * expf(lm_c[q][r] - M); Z += lz_c[q][r]; }
    part_m_c[bo + r] = M; part_s_c[bo + r] = S; part_z_c[bo + r] = Z;
  } else {
    const int r = tid - BATCH;
    float M = NEG_BIG;
#pragma unroll
    for (int q = 0; q < 8; ++q) M = fmaxf(M, lm_i[q][r]);
    float S = 0.f, Z = 0.f;
#pragma unroll
    for (int q = 0; q < 8; ++q) { S += ls_i[q][r] * expf(lm_i[q][r] - M); Z += lz_i[q][r]; }
    part_m_i[bo + r] = M; part_s_i[bo + r] = S; part_z_i[bo + r] = Z;
  }
  if (tid == 0) {
    float h = 0.f;
#pragma unroll
    for (int q = 0; q < 8; ++q) h += lh[q];
    part_h[blockIdx.x] = h;
  }
}

__global__ __launch_bounds__(256)
void dcc_final(const float* __restrict__ part_m_c, const float* __restrict__ part_s_c, const float* __restrict__ part_z_c,
               const float* __restrict__ part_m_i, const float* __restrict__ part_s_i, const float* __restrict__ part_z_i,
               const float* __restrict__ part_h,
               const float* __restrict__ zt_c, const float* __restrict__ zt_i,
               int NB, int C, float* __restrict__ out) {
  __shared__ float lloss[256];
  const int tid = threadIdx.x;
  float loss = 0.f;
  {
    const int r = tid & (BATCH - 1);
    const float* pm = (tid < BATCH) ? part_m_c : part_m_i;
    const float* ps = (tid < BATCH) ? part_s_c : part_s_i;
    const float* pz = (tid < BATCH) ? part_z_c : part_z_i;
    const float* zt = (tid < BATCH) ? zt_c : zt_i;
    float M = NEG_BIG;
    for (int i = 0; i < NB; ++i) M = fmaxf(M, pm[(size_t)i * BATCH + r]);
    double S = 0.0, Z = 0.0;
    for (int i = 0; i < NB; ++i) {
      S += (double)ps[(size_t)i * BATCH + r] * exp((double)pm[(size_t)i * BATCH + r] - (double)M);
      Z += (double)pz[(size_t)i * BATCH + r];
    }
    const float lse = M + (float)log(S);
    loss = lse - (1.f - LS_EPS) * zt[r] - (LS_EPS / (float)C) * (float)Z;
  }
  lloss[tid] = loss;
  __syncthreads();
  if (tid == 0) {
    double t = 0.0;
    for (int i = 0; i < 256; ++i) t += (double)lloss[i];
    double h = 0.0;
    for (int i = 0; i < NB; ++i) h += (double)part_h[i];
    const double total = t / (double)BATCH +
                         (double)WEIGHTV * h / ((double)BATCH * (double)C);
    out[0] = (float)total;
  }
}

extern "C" void kernel_launch(void* const* d_in, const int* in_sizes, int n_in,
                              void* d_out, int out_size, void* d_ws, size_t ws_size,
                              hipStream_t stream) {
  const float* inputs  = (const float*)d_in[0];
  const int*   targets = (const int*)d_in[1];
  const float* lut_ccc = (const float*)d_in[2];
  const float* lut_icc = (const float*)d_in[3];

  const int F  = FDIM;
  const int C  = in_sizes[2] / F;        // 100000
  const int NG = (C + 15) / 16;          // 16-class groups
  const int NB = (NG + 7) / 8;           // blocks (8 groups/block)

  float* ws = (float*)d_ws;
  const size_t n = (size_t)NB * BATCH;
  float* part_m_c = ws; ws += n;
  float* part_s_c = ws; ws += n;
  float* part_z_c = ws; ws += n;
  float* part_m_i = ws; ws += n;
  float* part_s_i = ws; ws += n;
  float* part_z_i = ws; ws += n;
  float* part_h   = ws; ws += NB;
  float* zt_c     = ws; ws += BATCH;
  float* zt_i     = ws; ws += BATCH;

  const size_t smem = (size_t)2 * AELEMS * sizeof(__hip_bfloat16); // 128KB A hi/lo planes

  dcc_gemm_partial<<<NB, 256, smem, stream>>>(inputs, targets, lut_ccc, lut_icc, C, NG,
                                              part_m_c, part_s_c, part_z_c,
                                              part_m_i, part_s_i, part_z_i,
                                              part_h, zt_c, zt_i);
  dcc_final<<<1, 256, 0, stream>>>(part_m_c, part_s_c, part_z_c,
                                   part_m_i, part_s_i, part_z_i,
                                   part_h, zt_c, zt_i, NB, C, (float*)d_out);
}
